// Glm4vVisionBlock_74457553043921
// MI455X (gfx1250) — matrix operand order repaired and checked by reading, not compiled, not run on hardware
//
#include <hip/hip_runtime.h>
#include <stdint.h>

// ---------------------------------------------------------------------------
// GLM4V vision block for MI455X (gfx1250, wave32, WMMA).
// GEMMs: bf16 WMMA 16x16x32, f32 accumulation, async global->LDS staging
// (GLOBAL_LOAD_ASYNC_TO_LDS_B128, ASYNCcnt) with 2-deep double buffering.
// ---------------------------------------------------------------------------

#define SEQ   3072
#define HID   1536
#define NH    12
#define HD    128
#define INTER 13696
#define QKV3  4608
#define EPSF  1e-6f

typedef __bf16 bf16;
typedef __attribute__((ext_vector_type(16))) __bf16 v16bf;
typedef __attribute__((ext_vector_type(8)))  __bf16 v8bf;
typedef __attribute__((ext_vector_type(8)))  float  v8f;

// ---- WMMA fragment loaders (layouts per CDNA5 ISA §7.12.2) -----
// A-matrix 16x32 bf16: lanes 0-15 hold row m, K=0..7 (v0-3) and K=16..23
// (v4-7); lanes 16-31 hold K=8..15 and K=24..31. Source: row-major, 16B rows.
__device__ __forceinline__ v16bf load_frag_a(const bf16* base, int ld) {
  int lane = threadIdx.x & 31;
  int r  = lane & 15;
  int kb = (lane & 16) ? 8 : 0;
  const bf16* p = base + r * ld + kb;
  union { v16bf v; v8bf h[2]; } u;
  u.h[0] = *(const v8bf*)(p);
  u.h[1] = *(const v8bf*)(p + 16);
  return u.v;
}

// B-matrix 32x16 bf16, loaded from B^T stored (N,K) row-major:
// lane n (0-15) = column n, holds K=0..7 and K=16..23; lanes 16-31 K=8..15 and K=24..31.
__device__ __forceinline__ v16bf load_frag_b(const bf16* base, int ld) {
  int lane = threadIdx.x & 31;
  int r  = lane & 15;
  int kb = (lane & 16) ? 8 : 0;
  const bf16* p = base + r * ld + kb;
  union { v16bf v; v8bf h[2]; } u;
  u.h[0] = *(const v8bf*)(p);
  u.h[1] = *(const v8bf*)(p + 16);
  return u.v;
}

__device__ __forceinline__ v8f wmma_bf16(v16bf a, v16bf b, v8f c) {
  return __builtin_amdgcn_wmma_f32_16x16x32_bf16(false, a, false, b,
                                                 (short)0, c, false, false);
}

// Async copy of 16B from global to LDS (per lane). Tracked by ASYNCcnt.
__device__ __forceinline__ void async_copy_b128(const bf16* gptr, bf16* lptr) {
  unsigned lds_addr = (unsigned)(uintptr_t)lptr;           // low 32 bits = LDS offset
  unsigned long long gaddr = (unsigned long long)(uintptr_t)gptr;
  asm volatile("global_load_async_to_lds_b128 %0, %1, off"
               :: "v"(lds_addr), "v"(gaddr) : "memory");
}

// ---------------------------------------------------------------------------
// fp32 (K,N) -> bf16 (N,K) transpose-cast (weights, done once per launch)
// ---------------------------------------------------------------------------
__global__ void cast_transpose_kernel(const float* __restrict__ src,
                                      bf16* __restrict__ dst,
                                      int K, int N) {
  long long idx = (long long)blockIdx.x * blockDim.x + threadIdx.x;
  long long total = (long long)K * N;
  if (idx >= total) return;
  int n = (int)(idx / K);
  int k = (int)(idx % K);
  dst[idx] = (bf16)src[(long long)k * N + n];
}

// ---------------------------------------------------------------------------
// RMSNorm: fp32 row -> bf16 row
// ---------------------------------------------------------------------------
__global__ void rmsnorm_kernel(const float* __restrict__ x,
                               const float* __restrict__ w,
                               bf16* __restrict__ out, int K) {
  int row = blockIdx.x;
  const float* xr = x + (long long)row * K;
  float s = 0.f;
  for (int i = threadIdx.x; i < K; i += blockDim.x) { float v = xr[i]; s += v * v; }
  __shared__ float red[256];
  red[threadIdx.x] = s;
  __syncthreads();
  for (int off = 128; off > 0; off >>= 1) {
    if ((int)threadIdx.x < off) red[threadIdx.x] += red[threadIdx.x + off];
    __syncthreads();
  }
  float inv = rsqrtf(red[0] / (float)K + EPSF);
  bf16* orow = out + (long long)row * K;
  for (int i = threadIdx.x; i < K; i += blockDim.x)
    orow[i] = (bf16)(xr[i] * inv * w[i]);
}

// ---------------------------------------------------------------------------
// bf16 WMMA GEMM: C(M,N) = A(M,K) x B(K,N), B given transposed (N,K).
// Block = 256 thr = 8 waves; tile 128x128; wave tile 64x32; K-step 32.
// Double-buffered LDS filled by GLOBAL_LOAD_ASYNC_TO_LDS_B128:
//   issue tile n+1 (4 async ops/thread) -> s_wait_asynccnt 4 (older tile done,
//   async loads retire in order) -> barrier -> WMMA on tile n -> barrier.
// Epilogues: 0 store f32, 1 +resid f32, 2 silu->bf16, 3 *mulb->bf16.
// All M,N,K here are multiples of 128/32 -> no bounds checks.
// ---------------------------------------------------------------------------
#define LDSS 40  // padded bf16 row stride (80B, keeps 16B alignment)

template <int EPI>
__global__ void __launch_bounds__(256)
gemm_bf16_kernel(const bf16* __restrict__ A, const bf16* __restrict__ Bt,
                 float* __restrict__ Cf, bf16* __restrict__ Cb,
                 const float* __restrict__ resid, const bf16* __restrict__ mulb,
                 int M, int N, int K) {
  __shared__ bf16 As[2][128 * LDSS];
  __shared__ bf16 Bs[2][128 * LDSS];
  int tid  = threadIdx.x;
  int lane = tid & 31;
  int wid  = tid >> 5;
  int wm = wid >> 2;   // 0..1 -> 64 rows
  int wn = wid & 3;    // 0..3 -> 32 cols
  int m0 = blockIdx.y * 128;
  int n0 = blockIdx.x * 128;

  // Per-thread staging slots: 2 x 16B per matrix per tile.
  int e0 = tid, e1 = tid + 256;
  int rowA0 = e0 >> 2, colA0 = (e0 & 3) * 8;
  int rowA1 = e1 >> 2, colA1 = (e1 & 3) * 8;

  auto stage = [&](int buf, int k0) {
    async_copy_b128(&A[(long long)(m0 + rowA0) * K + k0 + colA0],
                    &As[buf][rowA0 * LDSS + colA0]);
    async_copy_b128(&Bt[(long long)(n0 + rowA0) * K + k0 + colA0],
                    &Bs[buf][rowA0 * LDSS + colA0]);
    async_copy_b128(&A[(long long)(m0 + rowA1) * K + k0 + colA1],
                    &As[buf][rowA1 * LDSS + colA1]);
    async_copy_b128(&Bt[(long long)(n0 + rowA1) * K + k0 + colA1],
                    &Bs[buf][rowA1 * LDSS + colA1]);
  };

  v8f acc[4][2];
  for (int i = 0; i < 4; i++)
    for (int j = 0; j < 2; j++)
      for (int e = 0; e < 8; e++) acc[i][j][e] = 0.f;

  stage(0, 0);
  int it = 0;
  for (int k0 = 0; k0 < K; k0 += 32, it++) {
    int b = it & 1;
    bool have_next = (k0 + 32) < K;
    if (have_next) {
      stage(b ^ 1, k0 + 32);
      asm volatile("s_wait_asynccnt 0x4" ::: "memory");  // older tile landed
    } else {
      asm volatile("s_wait_asynccnt 0x0" ::: "memory");
    }
    __syncthreads();

    v16bf af[4], bfr[2];
    for (int i = 0; i < 4; i++)
      af[i] = load_frag_a(&As[b][(wm * 64 + i * 16) * LDSS], LDSS);
    for (int j = 0; j < 2; j++)
      bfr[j] = load_frag_b(&Bs[b][(wn * 32 + j * 16) * LDSS], LDSS);

    for (int i = 0; i < 4; i++)
      for (int j = 0; j < 2; j++)
        acc[i][j] = wmma_bf16(af[i], bfr[j], acc[i][j]);
    __syncthreads();  // reads done before buffer b is refilled next round
  }

  // Epilogue. C layout: vgpr e -> row e (lanes 0-15) / e+8 (lanes 16-31),
  // col = lane & 15.
  int half = (lane & 16) ? 8 : 0;
  int cn   = lane & 15;
  for (int i = 0; i < 4; i++)
    for (int j = 0; j < 2; j++) {
      int gcol = n0 + wn * 32 + j * 16 + cn;
      for (int r = 0; r < 8; r++) {
        int grow = m0 + wm * 64 + i * 16 + r + half;
        long long idx = (long long)grow * N + gcol;
        float v = acc[i][j][r];
        if (EPI == 0) {
          Cf[idx] = v;
        } else if (EPI == 1) {
          Cf[idx] = v + resid[idx];
        } else if (EPI == 2) {
          Cb[idx] = (bf16)(v / (1.f + __expf(-v)));     // SiLU
        } else {
          Cb[idx] = (bf16)(v * (float)mulb[idx]);
        }
      }
    }
}

// ---------------------------------------------------------------------------
// RoPE + repack: qkv fp32 (SEQ, 3*HID) ->
//   q,k bf16 (NH, SEQ, HD) row-major (A / B^T layouts for scores GEMM)
//   v  bf16 (NH, HD, SEQ)  (V^T, i.e. B^T layout for P·V GEMM)
// ---------------------------------------------------------------------------
__global__ void rope_kernel(const float* __restrict__ qkv,
                            const float* __restrict__ rot,
                            bf16* __restrict__ qb, bf16* __restrict__ kb,
                            bf16* __restrict__ vt) {
  int pos = blockIdx.x;
  int h   = blockIdx.y;
  int d   = threadIdx.x;  // 0..127
  float ang = rot[(long long)pos * HD + d];
  float c = cosf(ang), s = sinf(ang);
  const float* base = qkv + (long long)pos * QKV3;
  float q = base[h * HD + d];
  float k = base[HID + h * HD + d];
  float v = base[2 * HID + h * HD + d];
  int od = (d < HD / 2) ? d + HD / 2 : d - HD / 2;
  float sgn = (d < HD / 2) ? -1.f : 1.f;
  float qo = base[h * HD + od];
  float ko = base[HID + h * HD + od];
  qb[((long long)h * SEQ + pos) * HD + d] = (bf16)(q * c + sgn * qo * s);
  kb[((long long)h * SEQ + pos) * HD + d] = (bf16)(k * c + sgn * ko * s);
  vt[((long long)h * HD + d) * SEQ + pos] = (bf16)v;
}

// ---------------------------------------------------------------------------
// Flash attention, block-diagonal via cu_seqlens. One wave per 16 query rows
// per head. Segment lengths assumed multiples of 32 (1024 here).
// ---------------------------------------------------------------------------
__global__ void __launch_bounds__(128)
attn_kernel(const bf16* __restrict__ qb, const bf16* __restrict__ kb,
            const bf16* __restrict__ vt, const int* __restrict__ cu, int nseg,
            bf16* __restrict__ out) {
  __shared__ bf16 pstage[4][16 * 32];  // per-wave P staging (C->A relayout)
  int wid  = threadIdx.x >> 5;
  int lane = threadIdx.x & 31;
  int w = blockIdx.x * 4 + wid;
  const int nqt = SEQ / 16;  // 192
  int h  = w / nqt;
  int t  = w % nqt;
  int q0 = t * 16;

  int kbeg = 0, kend = SEQ;
  for (int i = 0; i < nseg; i++) {
    int s0 = cu[i], s1 = cu[i + 1];
    if (q0 >= s0 && q0 < s1) { kbeg = s0; kend = s1; }
  }

  const bf16* Q = qb + ((long long)h * SEQ + q0) * HD;
  v16bf qf[4];
  for (int kc = 0; kc < 4; kc++) qf[kc] = load_frag_a(Q + kc * 32, HD);

  v8f o[8];
  float mrow[8], lrow[8];
  for (int i = 0; i < 8; i++) {
    for (int e = 0; e < 8; e++) o[i][e] = 0.f;
    mrow[i] = -3.0e38f;
    lrow[i] = 0.f;
  }
  const float scale = 0.08838834764831845f;  // 1/sqrt(128)
  bf16* pst = &pstage[wid][0];

  for (int kk = kbeg; kk < kend; kk += 32) {
    v8f s0t, s1t;
    for (int e = 0; e < 8; e++) { s0t[e] = 0.f; s1t[e] = 0.f; }
    const bf16* Kp = kb + ((long long)h * SEQ + kk) * HD;
    if (kk + 32 < kend) {  // speculative prefetch of next K/V tiles
      __builtin_prefetch(Kp + 32 * HD + (lane << 3), 0, 1);
      __builtin_prefetch(vt + (long long)h * HD * SEQ + (long long)lane * 4 * SEQ + kk + 32, 0, 1);
    }
    for (int kc = 0; kc < 4; kc++) {
      v16bf kf = load_frag_b(Kp + kc * 32, HD);
      s0t = wmma_bf16(qf[kc], kf, s0t);
    }
    for (int kc = 0; kc < 4; kc++) {
      v16bf kf = load_frag_b(Kp + 16 * HD + kc * 32, HD);
      s1t = wmma_bf16(qf[kc], kf, s1t);
    }

    float nm[8], corr[8];
    for (int r = 0; r < 8; r++) {
      float a = s0t[r] * scale, b = s1t[r] * scale;
      s0t[r] = a; s1t[r] = b;
      float rm = fmaxf(a, b);
      for (int m = 1; m < 16; m <<= 1) rm = fmaxf(rm, __shfl_xor(rm, m, 32));
      nm[r]   = fmaxf(mrow[r], rm);
      corr[r] = __expf(mrow[r] - nm[r]);
      mrow[r] = nm[r];
    }
    for (int r = 0; r < 8; r++) {
      float p0 = __expf(s0t[r] - nm[r]);
      float p1 = __expf(s1t[r] - nm[r]);
      float rs = p0 + p1;
      for (int m = 1; m < 16; m <<= 1) rs += __shfl_xor(rs, m, 32);
      lrow[r] = lrow[r] * corr[r] + rs;
      s0t[r] = p0; s1t[r] = p1;
    }
    for (int nt = 0; nt < 8; nt++)
      for (int r = 0; r < 8; r++) o[nt][r] *= corr[r];

    // Relayout P (C layout) -> A layout via LDS (16 rows x 32 keys bf16).
    {
      int half = (lane & 16) ? 8 : 0;
      int col  = lane & 15;
      for (int r = 0; r < 8; r++) {
        pst[(r + half) * 32 + col]      = (bf16)s0t[r];
        pst[(r + half) * 32 + 16 + col] = (bf16)s1t[r];
      }
    }
    asm volatile("s_wait_dscnt 0x0" ::: "memory");
    v16bf pf = load_frag_a(pst, 32);

    const bf16* Vp = vt + (long long)h * HD * SEQ + kk;
    for (int nt = 0; nt < 8; nt++) {
      v16bf vf = load_frag_b(Vp + (long long)(nt * 16) * SEQ, SEQ);
      o[nt] = wmma_bf16(pf, vf, o[nt]);
    }
  }

  int half = (lane & 16) ? 8 : 0;
  int col  = lane & 15;
  for (int nt = 0; nt < 8; nt++)
    for (int r = 0; r < 8; r++) {
      float v  = o[nt][r] / lrow[r];
      int row  = q0 + r + half;
      out[(long long)row * HID + h * HD + nt * 16 + col] = (bf16)v;
    }
}

// ---------------------------------------------------------------------------
// Host orchestration
// ---------------------------------------------------------------------------
extern "C" void kernel_launch(void* const* d_in, const int* in_sizes, int n_in,
                              void* d_out, int out_size, void* d_ws, size_t ws_size,
                              hipStream_t stream) {
  const float* hidden = (const float*)d_in[0];
  const float* rot    = (const float*)d_in[1];
  const float* n1w    = (const float*)d_in[2];
  const float* n2w    = (const float*)d_in[3];
  const float* qkvw   = (const float*)d_in[4];
  const float* projw  = (const float*)d_in[5];
  const float* gatew  = (const float*)d_in[6];
  const float* upw    = (const float*)d_in[7];
  const float* downw  = (const float*)d_in[8];
  const int*   cu     = (const int*)d_in[9];
  int nseg = in_sizes[9] - 1;

  uint8_t* p = (uint8_t*)d_ws;
  auto alloc = [&](size_t bytes) {
    uint8_t* r = p;
    p += (bytes + 255) & ~(size_t)255;
    return r;
  };
  bf16*  qkvw_t = (bf16*)alloc((size_t)QKV3 * HID * 2);
  bf16*  projw_t= (bf16*)alloc((size_t)HID * HID * 2);
  bf16*  gatew_t= (bf16*)alloc((size_t)INTER * HID * 2);
  bf16*  upw_t  = (bf16*)alloc((size_t)INTER * HID * 2);
  bf16*  downw_t= (bf16*)alloc((size_t)HID * INTER * 2);
  bf16*  h1     = (bf16*)alloc((size_t)SEQ * HID * 2);
  float* qkvf   = (float*)alloc((size_t)SEQ * QKV3 * 4);
  bf16*  qbuf   = (bf16*)alloc((size_t)NH * SEQ * HD * 2);
  bf16*  kbuf   = (bf16*)alloc((size_t)NH * SEQ * HD * 2);
  bf16*  vtbuf  = (bf16*)alloc((size_t)NH * HD * SEQ * 2);
  bf16*  attnb  = (bf16*)alloc((size_t)SEQ * HID * 2);
  float* resid2 = (float*)alloc((size_t)SEQ * HID * 4);
  bf16*  h2     = (bf16*)alloc((size_t)SEQ * HID * 2);
  bf16*  silug  = (bf16*)alloc((size_t)SEQ * INTER * 2);
  bf16*  interb = (bf16*)alloc((size_t)SEQ * INTER * 2);

  auto xpose = [&](const float* s, bf16* d, int K, int N) {
    long long tot = (long long)K * N;
    cast_transpose_kernel<<<(unsigned)((tot + 255) / 256), 256, 0, stream>>>(s, d, K, N);
  };
  xpose(qkvw,  qkvw_t,  HID,   QKV3);
  xpose(projw, projw_t, HID,   HID);
  xpose(gatew, gatew_t, HID,   INTER);
  xpose(upw,   upw_t,   HID,   INTER);
  xpose(downw, downw_t, INTER, HID);

  // h1 = rmsnorm(hidden, norm1_w)
  rmsnorm_kernel<<<SEQ, 256, 0, stream>>>(hidden, n1w, h1, HID);

  // qkv = h1 @ qkv_w   (f32 out)
  gemm_bf16_kernel<0><<<dim3(QKV3 / 128, SEQ / 128), 256, 0, stream>>>(
      h1, qkvw_t, qkvf, nullptr, nullptr, nullptr, SEQ, QKV3, HID);

  // RoPE + repack
  rope_kernel<<<dim3(SEQ, NH), HD, 0, stream>>>(qkvf, rot, qbuf, kbuf, vtbuf);

  // Flash attention (2304 waves, 4 waves per block)
  attn_kernel<<<(NH * (SEQ / 16)) / 4, 128, 0, stream>>>(qbuf, kbuf, vtbuf, cu,
                                                         nseg, attnb);

  // resid2 = hidden + attn @ proj_w
  gemm_bf16_kernel<1><<<dim3(HID / 128, SEQ / 128), 256, 0, stream>>>(
      attnb, projw_t, resid2, nullptr, hidden, nullptr, SEQ, HID, HID);

  // h2 = rmsnorm(resid2, norm2_w)
  rmsnorm_kernel<<<SEQ, 256, 0, stream>>>(resid2, n2w, h2, HID);

  // silug = silu(h2 @ gate_w) -> bf16
  gemm_bf16_kernel<2><<<dim3(INTER / 128, SEQ / 128), 256, 0, stream>>>(
      h2, gatew_t, nullptr, silug, nullptr, nullptr, SEQ, INTER, HID);

  // interb = silug * (h2 @ up_w) -> bf16
  gemm_bf16_kernel<3><<<dim3(INTER / 128, SEQ / 128), 256, 0, stream>>>(
      h2, upw_t, nullptr, interb, nullptr, silug, SEQ, INTER, HID);

  // out = resid2 + interb @ down_w  (f32)
  gemm_bf16_kernel<1><<<dim3(HID / 128, SEQ / 128), 256, 0, stream>>>(
      interb, downw_t, (float*)d_out, nullptr, resid2, nullptr, SEQ, HID, INTER);

  (void)n_in; (void)out_size; (void)ws_size;
}
